// Attention_83202106458238
// MI455X (gfx1250) — matrix operand order repaired and checked by reading, not compiled, not run on hardware
//
#include <hip/hip_runtime.h>
#include <hip/hip_bf16.h>

// ---------------------------------------------------------------------------
// MI455X (gfx1250) multi-head attention: bf16 WMMA flash-attention +
// LDS-free WMMA GEMMs + async-to-LDS (ASYNCcnt) double-buffered staging.
// Shapes: B=2, T=2048, E=1024, H=16, I=64.
// ---------------------------------------------------------------------------

typedef __attribute__((ext_vector_type(16))) __bf16          v16bf;
typedef __attribute__((ext_vector_type(8)))  float           v8f;
typedef __attribute__((ext_vector_type(16))) unsigned short  v16us;
typedef __attribute__((ext_vector_type(8)))  unsigned short  v8us;

#define B_    2
#define T_    2048
#define E_    1024
#define H_    16
#define I_    64
#define M_    (B_ * T_)   // 4096

// round-to-nearest-even f32 -> bf16 bits
static __device__ __forceinline__ unsigned short f32_to_bf16(float f) {
  unsigned int u = __builtin_bit_cast(unsigned int, f);
  unsigned int lsb = (u >> 16) & 1u;
  u += 0x7fffu + lsb;
  return (unsigned short)(u >> 16);
}

// A-fragment 16x32 bf16 (contiguous K order, matches ld_b_frag):
//  lane<16 : elems 0..15 -> K=0..15
//  lane>=16: elems 0..15 -> K=16..31
// rowp = &row[lane%16] data, K-base applied by caller.
static __device__ __forceinline__ v16bf ld_a_frag(const unsigned short* rowp, int half) {
  const unsigned short* p = rowp + half * 16;
  v8us lo = *(const v8us*)(p);
  v8us hi = *(const v8us*)(p + 8);
  v16us u;
#pragma unroll
  for (int e = 0; e < 8; ++e) { u[e] = lo[e]; u[e + 8] = hi[e]; }
  return __builtin_bit_cast(v16bf, u);
}

// B-fragment 32x16 bf16: lane = col N (lane%16); lanes 0-15 K=0..15,
// lanes 16-31 K=16..31; colp = column's data contiguous in K.
static __device__ __forceinline__ v16bf ld_b_frag(const unsigned short* colp, int half) {
  v16us u = *(const v16us*)(colp + half * 16);
  return __builtin_bit_cast(v16bf, u);
}

static __device__ __forceinline__ v8f wmma_bf16(v16bf a, v16bf b, v8f c) {
  return __builtin_amdgcn_wmma_f32_16x16x32_bf16(false, a, false, b, (short)0, c,
                                                 false, false);
}

// CDNA5 async copy: GLOBAL_LOAD_ASYNC_TO_LDS_B128 (tracked by ASYNCcnt).
// lds_dst low 32 bits of generic pointer == LDS byte address (ISA 10.2 aperture).
static __device__ __forceinline__ void async_ld_b128(unsigned short* lds_dst,
                                                     const unsigned short* gsrc) {
  unsigned int loff = (unsigned int)(unsigned long long)lds_dst;
  asm volatile("global_load_async_to_lds_b128 %0, %1, off"
               :: "v"(loff), "v"(gsrc) : "memory");
}

// ---------------------------------------------------------------------------
// Kernel 1a: plain f32 -> bf16 conversion
// ---------------------------------------------------------------------------
__global__ void cvt_f32_bf16(const float* __restrict__ src,
                             unsigned short* __restrict__ dst, int n) {
  int i = blockIdx.x * blockDim.x + threadIdx.x;
  int stride = gridDim.x * blockDim.x;
  for (; i < n; i += stride) dst[i] = f32_to_bf16(src[i]);
}

// Kernel 1b: weight conversion with transpose: [H][E][I] f32 -> [H][I][E] bf16
__global__ void cvt_wT(const float* __restrict__ src,
                       unsigned short* __restrict__ dst) {
  int idx = blockIdx.x * blockDim.x + threadIdx.x;
  int stride = gridDim.x * blockDim.x;
  for (; idx < H_ * E_ * I_; idx += stride) {
    int i = idx % I_;
    int e = (idx / I_) % E_;
    int h = idx / (E_ * I_);
    dst[((size_t)h * I_ + i) * E_ + e] = f32_to_bf16(src[idx]);
  }
}

// ---------------------------------------------------------------------------
// Kernel 2: QKV projection, LDS-free (L2-resident operands).
// grid (M/128, H, 3), block 128; wave = 32 rows x 64 cols (8 WMMA / K-step).
// proj 0,1 -> q,k as [proj][h][m][i]; proj 2 -> v transposed [h][b][i][t].
// ---------------------------------------------------------------------------
__global__ __launch_bounds__(128) void qkv_proj(
    const unsigned short* __restrict__ xb,   // [M][E] bf16
    const unsigned short* __restrict__ wT,   // [3][H][I][E] bf16 (pre-transposed)
    unsigned short* __restrict__ qk,         // [2][H][M][I] bf16
    unsigned short* __restrict__ vT) {       // [H][B][I][T] bf16
  const int mt = blockIdx.x, h = blockIdx.y, proj = blockIdx.z;
  const int lane = threadIdx.x & 31, wave = threadIdx.x >> 5;
  const int half = lane >> 4, n16 = lane & 15;
  const int row0 = mt * 128 + wave * 32;
  const unsigned short* wbase = wT + ((size_t)(proj * H_ + h)) * I_ * E_;

  v8f acc[2][4] = {};
  for (int k0 = 0; k0 < E_; k0 += 32) {
    v16bf a0 = ld_a_frag(xb + (size_t)(row0 + n16) * E_ + k0, half);
    v16bf a1 = ld_a_frag(xb + (size_t)(row0 + 16 + n16) * E_ + k0, half);
#pragma unroll
    for (int nt = 0; nt < 4; ++nt) {
      v16bf b = ld_b_frag(wbase + (size_t)(nt * 16 + n16) * E_ + k0, half);
      acc[0][nt] = wmma_bf16(a0, b, acc[0][nt]);
      acc[1][nt] = wmma_bf16(a1, b, acc[1][nt]);
    }
  }

  if (proj < 2) {
    unsigned short* out = qk + ((size_t)(proj * H_ + h)) * M_ * I_;
#pragma unroll
    for (int a = 0; a < 2; ++a)
#pragma unroll
      for (int nt = 0; nt < 4; ++nt)
#pragma unroll
        for (int r = 0; r < 8; ++r) {
          int m = row0 + a * 16 + r + half * 8;
          out[(size_t)m * I_ + nt * 16 + n16] = f32_to_bf16(acc[a][nt][r]);
        }
  } else {
#pragma unroll
    for (int a = 0; a < 2; ++a)
#pragma unroll
      for (int nt = 0; nt < 4; ++nt)
#pragma unroll
        for (int r = 0; r < 8; ++r) {
          int m = row0 + a * 16 + r + half * 8;
          int b = m >> 11, t = m & (T_ - 1);
          int i = nt * 16 + n16;
          vT[(((size_t)h * B_ + b) * I_ + i) * T_ + t] = f32_to_bf16(acc[a][nt][r]);
        }
  }
}

// ---------------------------------------------------------------------------
// Kernel 3: flash attention. grid (T/64, B, H), block 128 (4 waves x 16 rows).
// K and V^T tiles staged via async-to-LDS, double buffered.
// ---------------------------------------------------------------------------
__global__ __launch_bounds__(128) void flash_attn(
    const unsigned short* __restrict__ q,    // [H][B][T][I] bf16
    const unsigned short* __restrict__ k,    // [H][B][T][I] bf16
    const unsigned short* __restrict__ vT,   // [H][B][I][T] bf16
    unsigned short* __restrict__ att) {      // [B*T][E] bf16
  __shared__ unsigned short sK[2][32 * 64];   // [key][i]
  __shared__ unsigned short sVt[2][64 * 32];  // [dim][key]
  __shared__ unsigned short sP[4][16 * 32];   // per-wave P tile

  const int mt = blockIdx.x, b = blockIdx.y, h = blockIdx.z;
  const int tid = threadIdx.x, lane = tid & 31, wave = tid >> 5;
  const int half = lane >> 4, n16 = lane & 15;
  const size_t hb = ((size_t)h * B_ + b) * T_;
  const unsigned short* vbase = vT + ((size_t)h * B_ + b) * I_ * T_;

  // Q A-fragments (K 0..31 / 32..63), read once from global.
  const unsigned short* qrow = q + (hb + mt * 64 + wave * 16 + n16) * I_;
  v16bf qa0 = ld_a_frag(qrow, half);
  v16bf qa1 = ld_a_frag(qrow + 32, half);

  // async tile issue: 4 wave-instructions per tile (2 for K, 2 for V^T)
  const int c0 = tid, c1 = tid + 128;  // 16B chunk ids, unconditional
  auto issue_tile = [&](int kt, int buf) {
    const unsigned short* kg = k + (hb + kt) * I_;
    async_ld_b128(&sK[buf][c0 * 8], kg + c0 * 8);
    async_ld_b128(&sK[buf][c1 * 8], kg + c1 * 8);
    int r0 = c0 >> 2, o0 = (c0 & 3) * 8;
    int r1 = c1 >> 2, o1 = (c1 & 3) * 8;
    async_ld_b128(&sVt[buf][r0 * 32 + o0], vbase + (size_t)r0 * T_ + kt + o0);
    async_ld_b128(&sVt[buf][r1 * 32 + o1], vbase + (size_t)r1 * T_ + kt + o1);
  };

  v8f acc[4] = {};
  float mrow[8], lrow[8];
#pragma unroll
  for (int r = 0; r < 8; ++r) { mrow[r] = -1e30f; lrow[r] = 0.f; }
  const float scale = 0.022097086912079608f;  // 1/sqrt(T)

  issue_tile(0, 0);
  for (int kt = 0; kt < T_; kt += 32) {
    const int cur = (kt >> 5) & 1;
    if (kt + 32 < T_) {
      issue_tile(kt + 32, cur ^ 1);
      __builtin_prefetch(k + (hb + kt + 64) * I_, 0, 0);  // global_prefetch
      asm volatile("s_wait_asynccnt 4" ::: "memory");     // current tile done
    } else {
      asm volatile("s_wait_asynccnt 0" ::: "memory");
    }
    __syncthreads();  // publish tile to all waves

    // S = Q * K^T : two 16-key groups, K-dim 64 -> 2 WMMA steps each
    v8f s0 = {}, s1 = {};
    {
      v16bf b00 = ld_b_frag(&sK[cur][(n16) * 64], half);
      v16bf b01 = ld_b_frag(&sK[cur][(n16) * 64 + 32], half);
      s0 = wmma_bf16(qa0, b00, s0);
      s0 = wmma_bf16(qa1, b01, s0);
      v16bf b10 = ld_b_frag(&sK[cur][(16 + n16) * 64], half);
      v16bf b11 = ld_b_frag(&sK[cur][(16 + n16) * 64 + 32], half);
      s1 = wmma_bf16(qa0, b10, s1);
      s1 = wmma_bf16(qa1, b11, s1);
    }

    // online softmax (row r lives across the 16 lanes of this half)
#pragma unroll
    for (int r = 0; r < 8; ++r) {
      float a0 = s0[r] * scale, a1 = s1[r] * scale;
      float mx = fmaxf(a0, a1);
#pragma unroll
      for (int d = 1; d < 16; d <<= 1) mx = fmaxf(mx, __shfl_xor(mx, d, 32));
      float mnew = fmaxf(mrow[r], mx);
      float corr = __expf(mrow[r] - mnew);
      float p0 = __expf(a0 - mnew), p1 = __expf(a1 - mnew);
      float ls = p0 + p1;
#pragma unroll
      for (int d = 1; d < 16; d <<= 1) ls += __shfl_xor(ls, d, 32);
      lrow[r] = lrow[r] * corr + ls;
      mrow[r] = mnew;
#pragma unroll
      for (int nt = 0; nt < 4; ++nt) acc[nt][r] *= corr;
      int m = r + half * 8;  // C-layout row
      sP[wave][m * 32 + n16] = f32_to_bf16(p0);
      sP[wave][m * 32 + 16 + n16] = f32_to_bf16(p1);
    }
    // per-wave LDS round trip: DS ops are in-order per wave; wait for writes
    asm volatile("s_wait_dscnt 0" ::: "memory");

    // acc += P(16x32) * V(32x64)
    v16bf pa = ld_a_frag(&sP[wave][n16 * 32], half);
#pragma unroll
    for (int nt = 0; nt < 4; ++nt) {
      v16bf bv = ld_b_frag(&sVt[cur][(nt * 16 + n16) * 32], half);
      acc[nt] = wmma_bf16(pa, bv, acc[nt]);
    }
    __syncthreads();  // all waves done reading buf before it is overwritten
  }

  // epilogue: attended[b][t][(h i)] = acc / l
  unsigned short* obase =
      att + ((size_t)(b * T_ + mt * 64 + wave * 16)) * E_ + h * I_;
#pragma unroll
  for (int r = 0; r < 8; ++r) {
    float inv = 1.0f / lrow[r];
    int m = r + half * 8;
#pragma unroll
    for (int nt = 0; nt < 4; ++nt)
      obase[(size_t)m * E_ + nt * 16 + n16] = f32_to_bf16(acc[nt][r] * inv);
  }
}

// ---------------------------------------------------------------------------
// Kernel 4: output projection (LDS-free). out[m][n] = sum_e att[m][e]*Wo[n][e]
// grid (M/128, E/64), block 128; wave = 32 rows x 64 cols.
// ---------------------------------------------------------------------------
__global__ __launch_bounds__(128) void out_proj(
    const unsigned short* __restrict__ att,  // [M][E] bf16
    const unsigned short* __restrict__ wo,   // [E][E] bf16 (row-major = B layout)
    float* __restrict__ out) {               // [M][E] f32
  const int mt = blockIdx.x, ntile = blockIdx.y;
  const int lane = threadIdx.x & 31, wave = threadIdx.x >> 5;
  const int half = lane >> 4, n16 = lane & 15;
  const int row0 = mt * 128 + wave * 32;

  v8f acc[2][4] = {};
  for (int k0 = 0; k0 < E_; k0 += 32) {
    v16bf a0 = ld_a_frag(att + (size_t)(row0 + n16) * E_ + k0, half);
    v16bf a1 = ld_a_frag(att + (size_t)(row0 + 16 + n16) * E_ + k0, half);
#pragma unroll
    for (int nt = 0; nt < 4; ++nt) {
      v16bf b = ld_b_frag(wo + (size_t)(ntile * 64 + nt * 16 + n16) * E_ + k0, half);
      acc[0][nt] = wmma_bf16(a0, b, acc[0][nt]);
      acc[1][nt] = wmma_bf16(a1, b, acc[1][nt]);
    }
  }

#pragma unroll
  for (int a = 0; a < 2; ++a)
#pragma unroll
    for (int nt = 0; nt < 4; ++nt)
#pragma unroll
      for (int r = 0; r < 8; ++r) {
        int m = row0 + a * 16 + r + half * 8;
        int n = ntile * 64 + nt * 16 + n16;
        out[(size_t)m * E_ + n] = acc[a][nt][r];
      }
}

// ---------------------------------------------------------------------------
// Host launcher
// ---------------------------------------------------------------------------
extern "C" void kernel_launch(void* const* d_in, const int* in_sizes, int n_in,
                              void* d_out, int out_size, void* d_ws, size_t ws_size,
                              hipStream_t stream) {
  const float* x  = (const float*)d_in[0];  // [B,T,E]
  const float* Wq = (const float*)d_in[1];  // [H,E,I]
  const float* Wk = (const float*)d_in[2];
  const float* Wv = (const float*)d_in[3];
  const float* Wo = (const float*)d_in[4];  // [E,E]
  float* out = (float*)d_out;

  unsigned short* ws = (unsigned short*)d_ws;
  const size_t NX   = (size_t)M_ * E_;        // 4,194,304
  const size_t NW   = (size_t)H_ * E_ * I_;   // 1,048,576
  const size_t NQKV = (size_t)H_ * M_ * I_;   // 4,194,304

  unsigned short* xb  = ws;                   // [M][E]
  unsigned short* wT  = xb + NX;              // [3][H][I][E]
  unsigned short* wob = wT + 3 * NW;          // [E][E]
  unsigned short* qb  = wob + (size_t)E_ * E_;// [2][H][M][I] : q then k
  unsigned short* kb  = qb + NQKV;
  unsigned short* vTb = kb + NQKV;            // [H][B][I][T]
  unsigned short* att = vTb + NQKV;           // [M][E]

  cvt_f32_bf16<<<2048, 256, 0, stream>>>(x, xb, (int)NX);
  cvt_wT<<<1024, 256, 0, stream>>>(Wq, wT + 0 * NW);
  cvt_wT<<<1024, 256, 0, stream>>>(Wk, wT + 1 * NW);
  cvt_wT<<<1024, 256, 0, stream>>>(Wv, wT + 2 * NW);
  cvt_f32_bf16<<<512, 256, 0, stream>>>(Wo, wob, E_ * E_);

  qkv_proj<<<dim3(M_ / 128, H_, 3), 128, 0, stream>>>(xb, wT, qb, vTb);
  flash_attn<<<dim3(T_ / 64, B_, H_), 128, 0, stream>>>(qb, kb, vTb, att);
  out_proj<<<dim3(M_ / 128, E_ / 64), 128, 0, stream>>>(att, wob, out);
}
